// GlobalLocalAttention_60249801228428
// MI455X (gfx1250) — matrix operand order repaired and checked by reading, not compiled, not run on hardware
//
#include <hip/hip_runtime.h>
#include <math.h>

typedef __bf16 bf16;
typedef __attribute__((ext_vector_type(8)))  __bf16 v8bf;
typedef __attribute__((ext_vector_type(16))) __bf16 v16bf;
typedef __attribute__((ext_vector_type(8)))  float  v8f;

__device__ __forceinline__ bf16 f2bf(float x) {
  union { float f; unsigned u; } v; v.f = x;
  unsigned r = v.u + 0x7FFFu + ((v.u >> 16) & 1u);
  union { unsigned short s; bf16 b; } o; o.s = (unsigned short)(r >> 16);
  return o.b;
}

__device__ __forceinline__ v16bf cat8(v8bf lo, v8bf hi) {
  return __builtin_shufflevector(lo, hi, 0, 1, 2, 3, 4, 5, 6, 7,
                                 8, 9, 10, 11, 12, 13, 14, 15);
}

// ---------------------------------------------------------------------------
// bf16 WMMA GEMM: C(MxN, f32 or bf16) = A(MxK) * B(KxN) [+ bias[n]]
// TA: A stored (K,M) phys (A[k*lda+m]); TB: B stored (N,K) phys (B[n*ldb+k]).
// Block: 256 threads = 8 waves; block tile 128(M) x 64(N); wave tile 32x32
// (2x2 accumulators -> 4 v_wmma_f32_16x16x32_bf16 per k-step). Fragments are
// ds_load_b128 pairs from 16B-aligned LDS rows (B staged transposed so both
// A and B fragments are contiguous per lane). Global->LDS staging uses 16B
// vector loads; all strides here are compile-time multiples of 8 bf16.
// ---------------------------------------------------------------------------
template <bool TA, bool TB, bool OUTBF, bool HASBIAS>
__global__ void __launch_bounds__(256)
gemm_wmma(const bf16* __restrict__ A, const bf16* __restrict__ Bm,
          void* __restrict__ Cout, const float* __restrict__ bias,
          int M, int N, int K, int lda, int ldb, int ldc) {
  __shared__ __align__(16) bf16 As[128][32];   // [m][k], row = 64B (16B aligned)
  __shared__ __align__(16) bf16 BsT[64][32];   // [n][k], transposed stage
  const int m0 = blockIdx.y * 128;
  const int n0 = blockIdx.x * 64;
  const int tid = threadIdx.x;
  const int wave = tid >> 5;
  const int lane = tid & 31;
  const int lh = lane & 15;        // low-half lane id
  const int hv = lane >> 4;        // high-half flag
  const int wm = (wave >> 1) << 5; // wave M offset: 0,32,64,96
  const int wn = (wave & 1) << 5;  // wave N offset: 0,32

  v8f acc00 = {0.f,0.f,0.f,0.f,0.f,0.f,0.f,0.f};
  v8f acc01 = acc00, acc10 = acc00, acc11 = acc00;

  for (int k0 = 0; k0 < K; k0 += 32) {
    // next-tile prefetch (emits global_prefetch_b8)
    if (k0 + 32 < K) {
      __builtin_prefetch(TA ? &A[(size_t)(k0 + 32 + (tid & 31)) * lda + m0]
                            : &A[(size_t)(m0 + (tid >> 1)) * lda + k0 + 32], 0, 1);
    }
    // ---- stage A (128x32): 512 x 16B chunks, 2 per thread ----
    if (!TA) {
#pragma unroll
      for (int e = tid; e < 512; e += 256) {
        int r = e >> 2, c8 = (e & 3) << 3;          // row, k-offset {0,8,16,24}
        *(v8bf*)&As[r][c8] =
            *(const v8bf*)&A[(size_t)(m0 + r) * lda + (k0 + c8)];
      }
    } else {
#pragma unroll
      for (int e = tid; e < 512; e += 256) {
        int k = e >> 4, r8 = (e & 15) << 3;          // k 0..31, m-offset step 8
        v8bf v = *(const v8bf*)&A[(size_t)(k0 + k) * lda + (m0 + r8)];
#pragma unroll
        for (int j = 0; j < 8; ++j) As[r8 + j][k] = v[j];
      }
    }
    // ---- stage B^T (64x32): 256 x 16B chunks, 1 per thread ----
    {
      int e = tid;
      if (TB) {
        int n = e >> 2, c8 = (e & 3) << 3;
        *(v8bf*)&BsT[n][c8] =
            *(const v8bf*)&Bm[(size_t)(n0 + n) * ldb + (k0 + c8)];
      } else {
        int k = e >> 3, n8 = (e & 7) << 3;           // k 0..31, n-offset step 8
        v8bf v = *(const v8bf*)&Bm[(size_t)(k0 + k) * ldb + (n0 + n8)];
#pragma unroll
        for (int j = 0; j < 8; ++j) BsT[n8 + j][k] = v[j];
      }
    }
    __syncthreads();

    // A fragments (M sub-tiles 0,16): lane layout per ISA 7.12.2 —
    // halfs [0..7] = K hv*8..+7, halfs [8..15] = K 16+hv*8..+7 (contiguous)
    const v8bf* a0 = (const v8bf*)&As[wm + lh][0];
    const v8bf* a1 = (const v8bf*)&As[wm + 16 + lh][0];
    v16bf af0 = cat8(a0[hv], a0[2 + hv]);
    v16bf af1 = cat8(a1[hv], a1[2 + hv]);
    // B fragments (N sub-tiles 0,16): halfs [0..7] = K hv*8..+7, halfs [8..15] = K 16+hv*8..+7
    const v8bf* b0 = (const v8bf*)&BsT[wn + lh][0];
    const v8bf* b1 = (const v8bf*)&BsT[wn + 16 + lh][0];
    v16bf bf0 = cat8(b0[hv], b0[2 + hv]);
    v16bf bf1 = cat8(b1[hv], b1[2 + hv]);

    acc00 = __builtin_amdgcn_wmma_f32_16x16x32_bf16(false, af0, false, bf0, (short)0, acc00, false, false);
    acc01 = __builtin_amdgcn_wmma_f32_16x16x32_bf16(false, af0, false, bf1, (short)0, acc01, false, false);
    acc10 = __builtin_amdgcn_wmma_f32_16x16x32_bf16(false, af1, false, bf0, (short)0, acc10, false, false);
    acc11 = __builtin_amdgcn_wmma_f32_16x16x32_bf16(false, af1, false, bf1, (short)0, acc11, false, false);
    __syncthreads();
  }

  // D layout: VGPR r -> M = r + 8*hv, N = lh
#pragma unroll
  for (int mi = 0; mi < 2; ++mi) {
#pragma unroll
    for (int ni = 0; ni < 2; ++ni) {
      const v8f& acc = mi == 0 ? (ni == 0 ? acc00 : acc01)
                               : (ni == 0 ? acc10 : acc11);
      int nc = n0 + wn + (ni << 4) + lh;
      float bv = HASBIAS ? bias[nc] : 0.0f;
#pragma unroll
      for (int r = 0; r < 8; ++r) {
        int mr = m0 + wm + (mi << 4) + (hv << 3) + r;
        float vo = acc[r] + bv;
        if (OUTBF) ((bf16*)Cout)[(size_t)mr * ldc + nc] = f2bf(vo);
        else       ((float*)Cout)[(size_t)mr * ldc + nc] = vo;
      }
    }
  }
}

// ---------------------------------------------------------------------------
// Elementwise / reduction kernels (fp32 VALU — bandwidth-trivial at 23.3 TB/s)
// ---------------------------------------------------------------------------

__global__ void __launch_bounds__(256)
k_downsample(const float* __restrict__ f, const float* __restrict__ b,
             float* __restrict__ fds, float* __restrict__ bds,
             bf16* __restrict__ fdsb, bf16* __restrict__ bdsb, int n) {
  int idx = blockIdx.x * 256 + threadIdx.x;
  if (idx >= n) return;                       // n = 2*64*2304
  int q = idx % 2304, bc = idx / 2304;        // bc = sample*64 + c
  int y = q / 48, x = q % 48;
  size_t src = (size_t)bc * 9216 + (size_t)(2 * y) * 96 + 2 * x;
  float fv = f[src], bv = b[src];
  fds[idx] = fv;  bds[idx] = bv;
  fdsb[idx] = f2bf(fv); bdsb[idx] = f2bf(bv);
}

__global__ void __launch_bounds__(256)
k_maskds(const float* __restrict__ mask, float* __restrict__ mds) {
  int q = blockIdx.x * 256 + threadIdx.x;
  if (q >= 2304) return;
  int y = q / 48, x = q % 48;
  mds[q] = mask[(size_t)(2 * y) * 96 + 2 * x];   // sample 0 only (mask[0:1])
}

__global__ void __launch_bounds__(256)
k_mm(const float* __restrict__ mds, float* __restrict__ mm) {
  int p = blockIdx.x * 256 + threadIdx.x;
  if (p >= 2304) return;
  int py = p / 48, px = p % 48;
  float s = 0.f;
  for (int dy = -1; dy <= 1; ++dy)
    for (int dx = -1; dx <= 1; ++dx) {
      int y = py + dy, x = px + dx;
      if (y >= 0 && y < 48 && x >= 0 && x < 48) s += mds[y * 48 + x];
    }
  mm[p] = (s == 0.f) ? 1.f : 0.f;
}

__global__ void __launch_bounds__(256)
k_cvt_bf16(const float* __restrict__ src, bf16* __restrict__ dst, int n) {
  int i = blockIdx.x * 256 + threadIdx.x;
  if (i < n) dst[i] = f2bf(src[i]);
}

// row softmax: attn[i][j] = softmax_j(sim[i][j] * mv[i]) -> bf16
__global__ void __launch_bounds__(256)
k_softmax_rows(const float* __restrict__ sim, const float* __restrict__ mv,
               bf16* __restrict__ attn) {
  const int N = 2304;
  int i = blockIdx.x, tid = threadIdx.x;
  float scale = mv[i];
  __shared__ float red[256];
  float loc[9]; float mx = -1e30f;
#pragma unroll
  for (int t = 0; t < 9; ++t) {
    loc[t] = sim[(size_t)i * N + t * 256 + tid] * scale;
    mx = fmaxf(mx, loc[t]);
  }
  red[tid] = mx; __syncthreads();
  for (int s = 128; s > 0; s >>= 1) {
    if (tid < s) red[tid] = fmaxf(red[tid], red[tid + s]);
    __syncthreads();
  }
  mx = red[0]; __syncthreads();
  float sum = 0.f;
#pragma unroll
  for (int t = 0; t < 9; ++t) { loc[t] = __expf(loc[t] - mx); sum += loc[t]; }
  red[tid] = sum; __syncthreads();
  for (int s = 128; s > 0; s >>= 1) {
    if (tid < s) red[tid] += red[tid + s];
    __syncthreads();
  }
  float inv = 1.f / red[0];
#pragma unroll
  for (int t = 0; t < 9; ++t)
    attn[(size_t)i * N + t * 256 + tid] = f2bf(loc[t] * inv);
}

// fp[c][q] = gamma*fds*mask + (1-mask)*outT[q][c]
__global__ void __launch_bounds__(256)
k_fp(const float* __restrict__ fds, const float* __restrict__ mv,
     const float* __restrict__ outT, const float* __restrict__ gamma,
     float* __restrict__ fp, int n) {
  int idx = blockIdx.x * 256 + threadIdx.x;
  if (idx >= n) return;                      // n = 64*2304
  int c = idx / 2304, q = idx % 2304;
  float m = mv[q];
  fp[idx] = gamma[0] * fds[idx] * m + (1.f - m) * outT[(size_t)q * 64 + c];
}

// im2col of fp (pad 1, 3x3): colfp[(c*9+ky*3+kx)][q] bf16
__global__ void __launch_bounds__(256)
k_colfp(const float* __restrict__ fp, bf16* __restrict__ colfp, int n) {
  int idx = blockIdx.x * 256 + threadIdx.x;
  if (idx >= n) return;                      // n = 576*2304
  int r = idx / 2304, q = idx % 2304;
  int c = r / 9, jj = r % 9;
  int y = q / 48 + jj / 3 - 1, x = q % 48 + jj % 3 - 1;
  float v = (y >= 0 && y < 48 && x >= 0 && x < 48) ? fp[(size_t)c * 2304 + y * 48 + x] : 0.f;
  colfp[idx] = f2bf(v);
}

// per-patch attention (pa) + kernel normalization -> wiN[p][c*9 + 3*ky+kx] bf16
__global__ void __launch_bounds__(128)
k_patch_attn(const float* __restrict__ fds, const float* __restrict__ bds,
             const float* __restrict__ mds,
             const float* __restrict__ wq, const float* __restrict__ bq,
             const float* __restrict__ wk, const float* __restrict__ bk,
             const float* __restrict__ wv, const float* __restrict__ bv,
             const float* __restrict__ gamma, bf16* __restrict__ wiN) {
  __shared__ float Xs[64][9], Bp[64][9], Qs[64][9], Ks[64][9], Vs[64][9], Os[64][9];
  __shared__ float mp[9], att[9][9], red[128];
  const int p = blockIdx.x, py = p / 48, px = p % 48;
  const int tid = threadIdx.x;
  if (tid < 9) {
    int y = py + tid / 3 - 1, x = px + tid % 3 - 1;
    mp[tid] = (y >= 0 && y < 48 && x >= 0 && x < 48) ? mds[y * 48 + x] : 0.f;
  }
  for (int e = tid; e < 576; e += 128) {
    int c = e / 9, j = e % 9;
    int y = py + j / 3 - 1, x = px + j % 3 - 1;
    bool in = (y >= 0 && y < 48 && x >= 0 && x < 48);
    Xs[c][j] = in ? fds[(size_t)c * 2304 + y * 48 + x] : 0.f;
    Bp[c][j] = in ? bds[(size_t)c * 2304 + y * 48 + x] : 0.f;
  }
  __syncthreads();
  for (int e = tid; e < 576; e += 128) {
    int o = e / 9, j = e % 9;
    float aq = bq[o], ak = bk[o], av = bv[o];
    for (int c = 0; c < 64; ++c) {
      aq += wq[o * 64 + c] * Xs[c][j];
      ak += wk[o * 64 + c] * Bp[c][j];
      av += wv[o * 64 + c] * Xs[c][j];
    }
    Qs[o][j] = aq; Ks[o][j] = ak; Vs[o][j] = av;
  }
  __syncthreads();
  if (tid < 81) {
    int i = tid / 9, j = tid % 9;
    float s = 0.f;
    for (int c = 0; c < 64; ++c) s += Qs[c][i] * Ks[c][j];
    att[i][j] = s * mp[i];
  }
  __syncthreads();
  if (tid < 9) {
    float mx = -1e30f;
    for (int j = 0; j < 9; ++j) mx = fmaxf(mx, att[tid][j]);
    float s = 0.f;
    for (int j = 0; j < 9; ++j) { float e = __expf(att[tid][j] - mx); att[tid][j] = e; s += e; }
    float inv = 1.f / s;
    for (int j = 0; j < 9; ++j) att[tid][j] *= inv;
  }
  __syncthreads();
  float ss = 0.f;
  for (int e = tid; e < 576; e += 128) {
    int c = e / 9, i = e % 9;
    float o = 0.f;
    for (int j = 0; j < 9; ++j) o += Vs[c][j] * att[i][j];
    float v = gamma[0] * Xs[c][i] * mp[i] + (1.f - mp[i]) * o;
    Os[c][i] = v; ss += v * v;
  }
  red[tid] = ss; __syncthreads();
  for (int s = 64; s > 0; s >>= 1) { if (tid < s) red[tid] += red[tid + s]; __syncthreads(); }
  float scale = 1.f / fmaxf(sqrtf(red[0]), 1e-4f);
  for (int e = tid; e < 576; e += 128)
    wiN[(size_t)p * 576 + e] = f2bf(Os[e / 9][e % 9] * scale);
}

// diagonal "fuse" conv (eye(3), same pad) on flat 2304x2304 score map
__global__ void __launch_bounds__(256)
k_fuse(const float* __restrict__ in, float* __restrict__ outp, int n) {
  const int N = 2304;
  int idx = blockIdx.x * 256 + threadIdx.x;
  if (idx >= n) return;
  int r = idx / N, c = idx % N;
  float s = in[idx];
  if (r > 0 && c > 0)         s += in[(size_t)(r - 1) * N + (c - 1)];
  if (r < N - 1 && c < N - 1) s += in[(size_t)(r + 1) * N + (c + 1)];
  outp[idx] = s;
}

// index permutation (y*48+x) <-> (x*48+y) on both axes (involution)
__global__ void __launch_bounds__(256)
k_perm(const float* __restrict__ in, float* __restrict__ outp, int n) {
  const int N = 2304;
  int idx = blockIdx.x * 256 + threadIdx.x;
  if (idx >= n) return;
  int a = idx / N, b = idx % N;
  int pa = (a % 48) * 48 + a / 48;
  int pb = (b % 48) * 48 + b / 48;
  outp[idx] = in[(size_t)pa * N + pb];
}

// column softmax over patches p with mask mm, SCALE=10 -> bf16 (kept (p,q))
__global__ void __launch_bounds__(256)
k_colsoftmax(const float* __restrict__ y, const float* __restrict__ mm,
             bf16* __restrict__ outp) {
  const int N = 2304;
  int q = blockIdx.x * 64 + threadIdx.x;   // blockDim = (64,4)
  int ty = threadIdx.y, tx = threadIdx.x;
  __shared__ float redv[4][64], colmax[64], colsum[64];
  float mx = -1e30f;
  for (int p = ty; p < N; p += 4) {
    float l = y[(size_t)p * N + q] * (mm[p] * 10.f);
    mx = fmaxf(mx, l);
  }
  redv[ty][tx] = mx; __syncthreads();
  if (ty == 0)
    colmax[tx] = fmaxf(fmaxf(redv[0][tx], redv[1][tx]), fmaxf(redv[2][tx], redv[3][tx]));
  __syncthreads();
  float cm = colmax[tx];
  float s = 0.f;
  for (int p = ty; p < N; p += 4)
    s += __expf(y[(size_t)p * N + q] * (mm[p] * 10.f) - cm);
  redv[ty][tx] = s; __syncthreads();
  if (ty == 0) colsum[tx] = redv[0][tx] + redv[1][tx] + redv[2][tx] + redv[3][tx];
  __syncthreads();
  float inv = 1.f / colsum[tx];
  for (int p = ty; p < N; p += 4) {
    float e = __expf(y[(size_t)p * N + q] * (mm[p] * 10.f) - cm) * inv * mm[p];
    outp[(size_t)p * N + q] = f2bf(e);
  }
}

// raw 4x4 stride-2 background patches (pad 1): rawW[p][c*16+ky*4+kx] bf16
__global__ void __launch_bounds__(256)
k_rawW(const float* __restrict__ bsrc, bf16* __restrict__ rawW, int n) {
  int idx = blockIdx.x * 256 + threadIdx.x;
  if (idx >= n) return;                    // n = 2304*1024
  int p = idx / 1024, r = idx % 1024;
  int c = r / 16, kk = r % 16;
  int y = 2 * (p / 48) + kk / 4 - 1, x = 2 * (p % 48) + kk % 4 - 1;
  float v = (y >= 0 && y < 96 && x >= 0 && x < 96) ? bsrc[(size_t)c * 9216 + y * 96 + x] : 0.f;
  rawW[idx] = f2bf(v);
}

// transposed-conv gather (stride 2, pad 1, 4x4) from colmat[q][c*16+k], /4
__global__ void __launch_bounds__(256)
k_col2im(const float* __restrict__ colmat, float* __restrict__ outp, int n) {
  int idx = blockIdx.x * 256 + threadIdx.x;
  if (idx >= n) return;                    // n = 64*96*96
  int c = idx / 9216, rem = idx % 9216;
  int Y = rem / 96, X = rem % 96;
  float s = 0.f;
  for (int ky = 0; ky < 4; ++ky) {
    int t = Y + 1 - ky;
    if (t < 0 || (t & 1)) continue;
    int qy = t >> 1; if (qy >= 48) continue;
    for (int kx = 0; kx < 4; ++kx) {
      int u = X + 1 - kx;
      if (u < 0 || (u & 1)) continue;
      int qx = u >> 1; if (qx >= 48) continue;
      s += colmat[(size_t)(qy * 48 + qx) * 1024 + c * 16 + ky * 4 + kx];
    }
  }
  outp[idx] = 0.25f * s;
}

// ---------------------------------------------------------------------------
extern "C" void kernel_launch(void* const* d_in, const int* in_sizes, int n_in,
                              void* d_out, int out_size, void* d_ws, size_t ws_size,
                              hipStream_t stream) {
  const float* f    = (const float*)d_in[0];
  const float* bin  = (const float*)d_in[1];
  const float* mask = (const float*)d_in[2];
  const float* fa_wq = (const float*)d_in[3];
  const float* fa_bq = (const float*)d_in[4];
  const float* fa_wk = (const float*)d_in[5];
  const float* fa_bk = (const float*)d_in[6];
  const float* fa_wv = (const float*)d_in[7];
  const float* fa_bv = (const float*)d_in[8];
  const float* fa_gamma = (const float*)d_in[9];
  const float* pa_wq = (const float*)d_in[10];
  const float* pa_bq = (const float*)d_in[11];
  const float* pa_wk = (const float*)d_in[12];
  const float* pa_bk = (const float*)d_in[13];
  const float* pa_wv = (const float*)d_in[14];
  const float* pa_bv = (const float*)d_in[15];
  const float* pa_gamma = (const float*)d_in[16];
  float* outp = (float*)d_out;

  const int N = 2304;                 // 48*48 pixels == patches
  const size_t CN = (size_t)64 * N;   // 147456

  char* base = (char*)d_ws;
  size_t off = 0;
  auto alloc = [&](size_t bytes) -> void* {
    off = (off + 255) & ~(size_t)255;
    void* p = base + off; off += bytes; return p;
  };
  float* fds   = (float*)alloc(2 * CN * 4);
  float* bds   = (float*)alloc(2 * CN * 4);
  bf16*  fdsb  = (bf16*) alloc(2 * CN * 2);
  bf16*  bdsb  = (bf16*) alloc(2 * CN * 2);
  float* mds   = (float*)alloc(N * 4);
  float* mm    = (float*)alloc(N * 4);
  bf16*  wqb   = (bf16*) alloc(4096 * 2);
  bf16*  wkb   = (bf16*) alloc(4096 * 2);
  bf16*  wvb   = (bf16*) alloc(4096 * 2);
  bf16*  Qt    = (bf16*) alloc(CN * 2);
  bf16*  Kt    = (bf16*) alloc(CN * 2);
  bf16*  Vt    = (bf16*) alloc(CN * 2);
  float* bufA  = (float*)alloc((size_t)N * N * 4);
  float* bufB  = (float*)alloc((size_t)N * N * 4);   // also holds sim
  bf16*  attn  = (bf16*) alloc((size_t)N * N * 2);
  bf16*  ysm   = (bf16*) alloc((size_t)N * N * 2);
  float* outT  = (float*)alloc(CN * 4);
  float* fpbuf = (float*)alloc(CN * 4);
  bf16*  colfp = (bf16*) alloc((size_t)576 * N * 2);
  bf16*  wiN   = (bf16*) alloc((size_t)N * 576 * 2);
  bf16*  rawW  = (bf16*) alloc((size_t)N * 1024 * 2);
  float* colmat= (float*)alloc((size_t)N * 1024 * 4);

  // ---- batch-shared prep ----
  k_downsample<<<(int)((2 * CN + 255) / 256), 256, 0, stream>>>(f, bin, fds, bds, fdsb, bdsb, (int)(2 * CN));
  k_maskds<<<9, 256, 0, stream>>>(mask, mds);
  k_mm<<<9, 256, 0, stream>>>(mds, mm);
  k_cvt_bf16<<<16, 256, 0, stream>>>(fa_wq, wqb, 4096);
  k_cvt_bf16<<<16, 256, 0, stream>>>(fa_wk, wkb, 4096);
  k_cvt_bf16<<<16, 256, 0, stream>>>(fa_wv, wvb, 4096);

  for (int s = 0; s < 2; ++s) {
    const float* fds_s = fds + (size_t)s * CN;
    const float* bds_s = bds + (size_t)s * CN;
    const bf16*  fdsb_s = fdsb + (size_t)s * CN;
    const bf16*  bdsb_s = bdsb + (size_t)s * CN;

    // Q/K/V 1x1 convs: (2304x64) = fds^T(64x2304,TA) * W^T(TB) + bias
    gemm_wmma<true, true, true, true><<<dim3(1, 18), 256, 0, stream>>>(
        fdsb_s, wqb, Qt, fa_bq, N, 64, 64, N, 64, 64);
    gemm_wmma<true, true, true, true><<<dim3(1, 18), 256, 0, stream>>>(
        bdsb_s, wkb, Kt, fa_bk, N, 64, 64, N, 64, 64);
    gemm_wmma<true, true, true, true><<<dim3(1, 18), 256, 0, stream>>>(
        fdsb_s, wvb, Vt, fa_bv, N, 64, 64, N, 64, 64);
    // sim = Qt * Kt^T  (2304x2304, K=64)
    gemm_wmma<false, true, false, false><<<dim3(36, 18), 256, 0, stream>>>(
        Qt, Kt, bufB, nullptr, N, N, 64, 64, 64, N);
    k_softmax_rows<<<N, 256, 0, stream>>>(bufB, mds, attn);
    // out^T = attn * Vt  (2304x64, K=2304)
    gemm_wmma<false, false, false, false><<<dim3(1, 18), 256, 0, stream>>>(
        attn, Vt, outT, nullptr, N, 64, N, N, 64, 64);
    k_fp<<<(int)((CN + 255) / 256), 256, 0, stream>>>(fds_s, mds, outT, fa_gamma, fpbuf, (int)CN);
    k_colfp<<<(576 * N + 255) / 256, 256, 0, stream>>>(fpbuf, colfp, 576 * N);
    // patch attention -> normalized correlation kernels
    k_patch_attn<<<N, 128, 0, stream>>>(fds_s, bds_s, mds,
                                        pa_wq, pa_bq, pa_wk, pa_bk, pa_wv, pa_bv,
                                        pa_gamma, wiN);
    // yi = wiN(2304x576) * colfp(576x2304)
    gemm_wmma<false, false, false, false><<<dim3(36, 18), 256, 0, stream>>>(
        wiN, colfp, bufA, nullptr, N, N, 576, 576, N, N);
    // fuse (diag conv) -> permute -> fuse -> permute back
    int nn = N * N;
    k_fuse<<<(nn + 255) / 256, 256, 0, stream>>>(bufA, bufB, nn);
    k_perm<<<(nn + 255) / 256, 256, 0, stream>>>(bufB, bufA, nn);
    k_fuse<<<(nn + 255) / 256, 256, 0, stream>>>(bufA, bufB, nn);
    k_perm<<<(nn + 255) / 256, 256, 0, stream>>>(bufB, bufA, nn);
    // masked channel softmax over patches
    k_colsoftmax<<<36, dim3(64, 4), 0, stream>>>(bufA, mm, ysm);
    // raw background 4x4 patches, then deconv GEMM + col2im
    k_rawW<<<(N * 1024 + 255) / 256, 256, 0, stream>>>(bin + (size_t)s * 64 * 9216, rawW, N * 1024);
    gemm_wmma<true, false, false, false><<<dim3(16, 18), 256, 0, stream>>>(
        ysm, rawW, colmat, nullptr, N, 1024, N, N, 1024, 1024);
    k_col2im<<<(64 * 9216 + 255) / 256, 256, 0, stream>>>(colmat, outp + (size_t)s * 64 * 9216, 64 * 9216);
  }
}
